// MultiHeadSelfAttention_10771777978877
// MI455X (gfx1250) — matrix operand order repaired and checked by reading, not compiled, not run on hardware
//
#include <hip/hip_runtime.h>
#include <hip/hip_bf16.h>
#include <math.h>
#include <stdint.h>

// ---------------------------------------------------------------------------
// Causal multi-head self-attention for MI455X (gfx1250), wave32 + WMMA bf16.
//  - fp32 -> bf16 staging (weights pre-transposed to N-major)
//  - projection GEMMs: 32x64 register-blocked per wave (8 wmma / 32-k-step,
//    ~130 VGPRs -> no scratch spills)
//  - flash-style attention: wmma S=QK^T, online softmax, async V staging to
//    LDS (global_load_async_to_lds_b128 + s_wait_asynccnt), wmma O=PV
// ---------------------------------------------------------------------------

typedef __bf16 bf16_t;
typedef __bf16 v16bf  __attribute__((ext_vector_type(16)));
typedef __bf16 bf16x8 __attribute__((ext_vector_type(8)));
typedef float  v8f    __attribute__((ext_vector_type(8)));

#define D_MODEL   1024
#define NUM_HEADS 16
#define D_KEY     64
#define B_SZ      2
#define T_SZ      2048
#define M_TOK     (B_SZ * T_SZ)              // 4096 tokens
#define N_QKV     (3 * NUM_HEADS * D_KEY)    // 3072

// ---------------- fragment helpers (CDNA5 16x16x32 bf16 WMMA) ---------------

__device__ __forceinline__ v16bf join8(bf16x8 lo, bf16x8 hi) {
  v16bf r;
#pragma unroll
  for (int i = 0; i < 8; i++) { r[i] = lo[i]; r[i + 8] = hi[i]; }
  return r;
}

// A fragment: 16x32 (MxK), row-major source with leading dim ld.
__device__ __forceinline__ v16bf load_a_frag(const bf16_t* base, int ld,
                                             int m0, int k0, int lane) {
  int row = m0 + (lane & 15);
  int kb  = (lane >> 4) * 8;
  const bf16_t* p = base + (size_t)row * ld + k0 + kb;
  bf16x8 lo = *(const bf16x8*)p;
  bf16x8 hi = *(const bf16x8*)(p + 16);
  return join8(lo, hi);
}

// B fragment: 32x16 (KxN) from an N-major source: src[n * ld + k].
__device__ __forceinline__ v16bf load_b_frag_t(const bf16_t* base, int ld,
                                               int n0, int k0, int lane) {
  int col = n0 + (lane & 15);
  int kb  = (lane >> 4) * 8;
  const bf16_t* p = base + (size_t)col * ld + k0 + kb;
  bf16x8 lo = *(const bf16x8*)p;
  bf16x8 hi = *(const bf16x8*)(p + 16);
  return join8(lo, hi);
}

__device__ __forceinline__ v8f wmma_bf16(v16bf a, v16bf b, v8f c) {
  return __builtin_amdgcn_wmma_f32_16x16x32_bf16(
      false, a, false, b, (short)0, c, false, false);
}

// ---------------------------- conversion kernels ----------------------------

__global__ void k_cvt(const float* __restrict__ src, bf16_t* __restrict__ dst, int n) {
  int i = blockIdx.x * blockDim.x + threadIdx.x;
  if (i < n) dst[i] = (bf16_t)src[i];
}

// src [rows, cols] row-major fp32 -> dst [cols, rows] bf16 (N-major weights)
__global__ void k_cvt_t(const float* __restrict__ src, bf16_t* __restrict__ dst,
                        int rows, int cols) {
  int i = blockIdx.x * blockDim.x + threadIdx.x;
  if (i < rows * cols) {
    int r = i / cols, c = i - r * cols;
    dst[(size_t)c * rows + r] = (bf16_t)src[i];
  }
}

// ------------------------------- QKV GEMM -----------------------------------
// 32x64 output tile per wave: 2 A-frags x 4 B-frags -> 8 wmma per k-step.
// Epilogue scatters into Q [B,H,T,Dk] (pre-scaled 1/sqrt(Dk)), K [B,H,T,Dk],
// V transposed [B,H,Dk,T].

__global__ void __launch_bounds__(256, 1)
k_gemm_qkv(const bf16_t* __restrict__ X,
           const bf16_t* __restrict__ Wt,   // [3072,1024] N-major
           const float*  __restrict__ bias, // [3072]
           bf16_t* __restrict__ Q, bf16_t* __restrict__ K,
           bf16_t* __restrict__ Vt) {
  int lane = threadIdx.x & 31;
  int wid  = (blockIdx.x * blockDim.x + threadIdx.x) >> 5;
  const int ntiles = N_QKV / 64;                 // 48
  int tm = wid / ntiles, tn = wid - tm * ntiles;
  if (tm >= M_TOK / 32) return;
  int m0 = tm * 32, n0 = tn * 64;

  v8f acc[2][4];
  v8f zero = {0.f, 0.f, 0.f, 0.f, 0.f, 0.f, 0.f, 0.f};
#pragma unroll
  for (int i = 0; i < 2; i++)
#pragma unroll
    for (int j = 0; j < 4; j++) acc[i][j] = zero;

#pragma unroll 1
  for (int k0 = 0; k0 < D_MODEL; k0 += 32) {
    v16bf a[2], b[4];
#pragma unroll
    for (int i = 0; i < 2; i++) a[i] = load_a_frag(X, D_MODEL, m0 + 16 * i, k0, lane);
#pragma unroll
    for (int j = 0; j < 4; j++) b[j] = load_b_frag_t(Wt, D_MODEL, n0 + 16 * j, k0, lane);
#pragma unroll
    for (int i = 0; i < 2; i++)
#pragma unroll
      for (int j = 0; j < 4; j++) acc[i][j] = wmma_bf16(a[i], b[j], acc[i][j]);
  }

  int hi = lane >> 4, nl = lane & 15;
#pragma unroll
  for (int j = 0; j < 4; j++) {
    int colb = n0 + 16 * j + nl;
    float bv = bias[colb];
    int s = colb >> 10;          // 0=Q, 1=K, 2=V
    int h = (colb >> 6) & 15;
    int d = colb & 63;
#pragma unroll
    for (int i = 0; i < 2; i++) {
#pragma unroll
      for (int r = 0; r < 8; r++) {
        int row = m0 + 16 * i + r + 8 * hi;   // global token index
        int b_  = row >> 11;                  // / T_SZ
        int t   = row & (T_SZ - 1);
        int bh  = b_ * NUM_HEADS + h;
        float v = acc[i][j][r] + bv;
        if (s == 0)
          Q[((size_t)bh * T_SZ + t) * D_KEY + d] = (bf16_t)(v * 0.125f);
        else if (s == 1)
          K[((size_t)bh * T_SZ + t) * D_KEY + d] = (bf16_t)v;
        else
          Vt[((size_t)bh * D_KEY + d) * T_SZ + t] = (bf16_t)v;
      }
    }
  }
}

// ------------------------- flash attention kernel ---------------------------
// One wave per 16-row q-tile of one (b,h); 32-key blocks, online softmax.
// V tiles are staged to per-wave LDS with async global->LDS b128 transfers.

struct SmWave {
  bf16_t p[16 * 32];   // P staging (C-frag -> A-frag relayout), 1 KB
  bf16_t v[64 * 32];   // V tile [d][key], 4 KB
};

__global__ void __launch_bounds__(256, 1)
k_attn(const bf16_t* __restrict__ Q, const bf16_t* __restrict__ K,
       const bf16_t* __restrict__ Vt, bf16_t* __restrict__ O) {
  __shared__ __align__(16) SmWave sm[8];
  int lane  = threadIdx.x & 31;
  int wslot = threadIdx.x >> 5;
  int wid   = (blockIdx.x * blockDim.x + threadIdx.x) >> 5;
  const int qtiles = T_SZ / 16;                      // 128
  int qt = wid % qtiles;
  int bh = wid / qtiles;
  if (bh >= B_SZ * NUM_HEADS) return;                // wave-uniform
  int q0 = qt * 16;

  const bf16_t* Qbh = Q  + (size_t)bh * T_SZ * D_KEY;
  const bf16_t* Kbh = K  + (size_t)bh * T_SZ * D_KEY;
  const bf16_t* Vbh = Vt + (size_t)bh * D_KEY * T_SZ;

  v16bf qa0 = load_a_frag(Qbh, D_KEY, q0, 0,  lane);
  v16bf qa1 = load_a_frag(Qbh, D_KEY, q0, 32, lane);

  int hi = lane >> 4;
  int nl = lane & 15;

  float m[8], l[8];
  v8f o[4];
  v8f zero = {0.f, 0.f, 0.f, 0.f, 0.f, 0.f, 0.f, 0.f};
#pragma unroll
  for (int r = 0; r < 8; r++) { m[r] = -1e30f; l[r] = 0.f; }
#pragma unroll
  for (int t = 0; t < 4; t++) o[t] = zero;

  bf16_t* pl = sm[wslot].p;
  bf16_t* vl = sm[wslot].v;
  uint32_t vbase = (uint32_t)(uintptr_t)vl;   // LDS aperture: offset in addr[31:0]

#pragma unroll 1
  for (int k0 = 0; k0 <= q0 + 15; k0 += 32) {
    // ---- async-stage V tile [64 d x 32 keys] into per-wave LDS ----
    // previous iteration's ds reads of vl must drain before overwrite
    asm volatile("s_wait_dscnt 0" ::: "memory");
#pragma unroll
    for (int i = 0; i < 8; i++) {
      int idx   = i * 32 + lane;          // 256 b128 transfers total
      int row   = idx >> 2;               // d (0..63)
      int chunk = idx & 3;                // 8 keys per 16B chunk
      const bf16_t* src = Vbh + (size_t)row * T_SZ + k0 + chunk * 8;
      uint64_t ga = (uint64_t)(uintptr_t)src;
      uint32_t ld = vbase + (uint32_t)(row * 64 + chunk * 16);
      asm volatile("global_load_async_to_lds_b128 %0, %1, off"
                   :: "v"(ld), "v"(ga) : "memory");
    }
    // prefetch next K tile while we crunch this block
    __builtin_prefetch(Kbh + (size_t)(k0 + 32) * D_KEY, 0, 0);

    float s0[8], s1[8];
    // ---- S = Q K^T for two 16-key subtiles ----
#pragma unroll
    for (int kt = 0; kt < 2; kt++) {
      int keyb = k0 + kt * 16;
      v16bf kb0 = load_b_frag_t(Kbh, D_KEY, keyb, 0,  lane);
      v16bf kb1 = load_b_frag_t(Kbh, D_KEY, keyb, 32, lane);
      v8f c = zero;
      c = wmma_bf16(qa0, kb0, c);
      c = wmma_bf16(qa1, kb1, c);
      int key = keyb + nl;
      float* dst = kt ? s1 : s0;
#pragma unroll
      for (int r = 0; r < 8; r++) {
        int qrow = q0 + r + 8 * hi;
        dst[r] = (key <= qrow) ? c[r] : -1e30f;     // causal mask
      }
    }
    // ---- online softmax (row = 16 lanes within each half-wave) ----
    float scale[8];
#pragma unroll
    for (int r = 0; r < 8; r++) {
      float mx = fmaxf(s0[r], s1[r]);
#pragma unroll
      for (int sh = 1; sh < 16; sh <<= 1) mx = fmaxf(mx, __shfl_xor(mx, sh, 32));
      float mnew = fmaxf(m[r], mx);
      float p0 = __expf(s0[r] - mnew);
      float p1 = __expf(s1[r] - mnew);
      float sum = p0 + p1;
#pragma unroll
      for (int sh = 1; sh < 16; sh <<= 1) sum += __shfl_xor(sum, sh, 32);
      scale[r] = __expf(m[r] - mnew);
      l[r] = l[r] * scale[r] + sum;
      m[r] = mnew;
      s0[r] = p0; s1[r] = p1;
    }
    // ---- P: C-fragment -> LDS (row-major 16x32 bf16), rescale O ----
#pragma unroll
    for (int r = 0; r < 8; r++) {
      int row = r + 8 * hi;
      pl[row * 32 + nl]      = (bf16_t)s0[r];
      pl[row * 32 + 16 + nl] = (bf16_t)s1[r];
#pragma unroll
      for (int t = 0; t < 4; t++) o[t][r] *= scale[r];
    }
    asm volatile("s_wait_dscnt 0" ::: "memory");      // P store -> load fence
    asm volatile("s_wait_asynccnt 0x0" ::: "memory"); // V tile landed in LDS
    // ---- O += P V  (P A-frag from LDS; V B-frags from LDS tile) ----
    v16bf pa = load_a_frag(pl, 32, 0, 0, lane);
#pragma unroll
    for (int nt = 0; nt < 4; nt++) {
      v16bf vb = load_b_frag_t(vl, 32, nt * 16, 0, lane);
      o[nt] = wmma_bf16(pa, vb, o[nt]);
    }
  }

  // ---- normalize and write O -> [token, H*Dk] bf16 ----
  int b_ = bh / NUM_HEADS;
  int h_ = bh - b_ * NUM_HEADS;
#pragma unroll
  for (int r = 0; r < 8; r++) {
    float inv = 1.0f / l[r];
    int tok = b_ * T_SZ + q0 + r + 8 * hi;
#pragma unroll
    for (int nt = 0; nt < 4; nt++) {
      O[(size_t)tok * D_MODEL + h_ * D_KEY + nt * 16 + nl] =
          (bf16_t)(o[nt][r] * inv);
    }
  }
}

// ------------------------------ output GEMM ---------------------------------
// 32x64 output tile per wave.

__global__ void __launch_bounds__(256, 1)
k_gemm_out(const bf16_t* __restrict__ A,   // [M_TOK,1024] bf16
           const bf16_t* __restrict__ Wt,  // [1024,1024] N-major
           const float*  __restrict__ bias,
           float* __restrict__ out) {
  int lane = threadIdx.x & 31;
  int wid  = (blockIdx.x * blockDim.x + threadIdx.x) >> 5;
  const int ntiles = D_MODEL / 64;                 // 16
  int tm = wid / ntiles, tn = wid - tm * ntiles;
  if (tm >= M_TOK / 32) return;
  int m0 = tm * 32, n0 = tn * 64;

  v8f acc[2][4];
  v8f zero = {0.f, 0.f, 0.f, 0.f, 0.f, 0.f, 0.f, 0.f};
#pragma unroll
  for (int i = 0; i < 2; i++)
#pragma unroll
    for (int j = 0; j < 4; j++) acc[i][j] = zero;

#pragma unroll 1
  for (int k0 = 0; k0 < D_MODEL; k0 += 32) {
    v16bf a[2], b[4];
#pragma unroll
    for (int i = 0; i < 2; i++) a[i] = load_a_frag(A, D_MODEL, m0 + 16 * i, k0, lane);
#pragma unroll
    for (int j = 0; j < 4; j++) b[j] = load_b_frag_t(Wt, D_MODEL, n0 + 16 * j, k0, lane);
#pragma unroll
    for (int i = 0; i < 2; i++)
#pragma unroll
      for (int j = 0; j < 4; j++) acc[i][j] = wmma_bf16(a[i], b[j], acc[i][j]);
  }

  int hi = lane >> 4, nl = lane & 15;
#pragma unroll
  for (int j = 0; j < 4; j++) {
    float bv = bias[n0 + 16 * j + nl];
#pragma unroll
    for (int i = 0; i < 2; i++) {
#pragma unroll
      for (int r = 0; r < 8; r++) {
        out[(size_t)(m0 + 16 * i + r + 8 * hi) * D_MODEL + n0 + 16 * j + nl] =
            acc[i][j][r] + bv;
      }
    }
  }
}

// ------------------------------- launcher -----------------------------------

extern "C" void kernel_launch(void* const* d_in, const int* in_sizes, int n_in,
                              void* d_out, int out_size, void* d_ws, size_t ws_size,
                              hipStream_t stream) {
  (void)in_sizes; (void)n_in; (void)out_size; (void)ws_size;
  const float* X    = (const float*)d_in[0];
  const float* Wqkv = (const float*)d_in[1];
  const float* bqkv = (const float*)d_in[2];
  const float* Wout = (const float*)d_in[3];
  const float* bout = (const float*)d_in[4];
  float* out = (float*)d_out;

  char* ws = (char*)d_ws;
  bf16_t* Xb    = (bf16_t*)ws; ws += (size_t)M_TOK * D_MODEL * 2;   //  8 MB
  bf16_t* WqkvT = (bf16_t*)ws; ws += (size_t)N_QKV * D_MODEL * 2;   //  6 MB
  bf16_t* WoutT = (bf16_t*)ws; ws += (size_t)D_MODEL * D_MODEL * 2; //  2 MB
  bf16_t* Qb    = (bf16_t*)ws; ws += (size_t)M_TOK * D_MODEL * 2;   //  8 MB
  bf16_t* Kb    = (bf16_t*)ws; ws += (size_t)M_TOK * D_MODEL * 2;   //  8 MB
  bf16_t* Vt    = (bf16_t*)ws; ws += (size_t)M_TOK * D_MODEL * 2;   //  8 MB
  bf16_t* Ao    = (bf16_t*)ws;                                      //  8 MB

  k_cvt  <<<(M_TOK * D_MODEL + 255) / 256, 256, 0, stream>>>(X, Xb, M_TOK * D_MODEL);
  k_cvt_t<<<(D_MODEL * N_QKV + 255) / 256, 256, 0, stream>>>(Wqkv, WqkvT, D_MODEL, N_QKV);
  k_cvt_t<<<(D_MODEL * D_MODEL + 255) / 256, 256, 0, stream>>>(Wout, WoutT, D_MODEL, D_MODEL);

  // QKV projection: (4096/32)*(3072/64) = 6144 wave-tiles, 8 waves/block
  k_gemm_qkv<<<6144 / 8, 256, 0, stream>>>(Xb, WqkvT, bqkv, Qb, Kb, Vt);

  // attention: B*H*(T/16) = 4096 waves, 8 waves/block
  k_attn<<<4096 / 8, 256, 0, stream>>>(Qb, Kb, Vt, Ao);

  // output projection: (4096/32)*(1024/64) = 2048 wave-tiles
  k_gemm_out<<<2048 / 8, 256, 0, stream>>>(Ao, WoutT, bout, out);
}
